// MaxMarginLoss_60868276519355
// MI455X (gfx1250) — compile-verified, matrix operand order checked by reading, not run on hardware
//
#include <hip/hip_runtime.h>
#include <hip/hip_bf16.h>

typedef __attribute__((ext_vector_type(16))) _Float16 v16h;
typedef __attribute__((ext_vector_type(8)))  float    v8f;
typedef __attribute__((ext_vector_type(4)))  _Float16 h4;

#define EPS_F 1e-8f
#define NEG_INF_F (-3.0e38f)

union AFrag { v16h h; float4 f4[2]; };

// ---------------------------------------------------------------------------
// Kernel 0: zero the scalar output (d_out is poisoned by the harness)
// ---------------------------------------------------------------------------
__global__ void mm_zero_kernel(float* out) { out[0] = 0.0f; }

// ---------------------------------------------------------------------------
// Kernel 1: normalize rows (veclist -> f16), store 1/norm. One wave per row.
// ---------------------------------------------------------------------------
__global__ void __launch_bounds__(256)
mm_normalize_rows_kernel(const float* __restrict__ src, _Float16* __restrict__ dst,
                         float* __restrict__ inv_norm, int rows) {
    const int row  = blockIdx.x * 8 + (threadIdx.x >> 5);
    const int lane = threadIdx.x & 31;
    if (row >= rows) return;

    const float* r = src + (size_t)row * 512;
    float4 v[4];
    float s = 0.0f;
#pragma unroll
    for (int t = 0; t < 4; ++t) {
        v[t] = *(const float4*)(r + t * 128 + lane * 4);
        s += v[t].x * v[t].x + v[t].y * v[t].y + v[t].z * v[t].z + v[t].w * v[t].w;
    }
#pragma unroll
    for (int mask = 16; mask >= 1; mask >>= 1) s += __shfl_xor(s, mask);

    const float inv = 1.0f / fmaxf(sqrtf(s), EPS_F);
    if (lane == 0) inv_norm[row] = inv;

    _Float16* o = dst + (size_t)row * 512;
#pragma unroll
    for (int t = 0; t < 4; ++t) {
        h4 h;
        h.x = (_Float16)(v[t].x * inv);
        h.y = (_Float16)(v[t].y * inv);
        h.z = (_Float16)(v[t].z * inv);
        h.w = (_Float16)(v[t].w * inv);
        *(h4*)(o + t * 128 + lane * 4) = h;
    }
}

// ---------------------------------------------------------------------------
// Kernel 2: normalize input rows -> f16, compute d_pos and 1/||input||.
// ---------------------------------------------------------------------------
__global__ void __launch_bounds__(256)
mm_prep_input_kernel(const float* __restrict__ input, const float* __restrict__ target,
                     _Float16* __restrict__ inn, float* __restrict__ in_inv,
                     float* __restrict__ d_pos, int B) {
    const int row  = blockIdx.x * 8 + (threadIdx.x >> 5);
    const int lane = threadIdx.x & 31;
    if (row >= B) return;

    const float* ri = input  + (size_t)row * 512;
    const float* rt = target + (size_t)row * 512;
    float4 vi[4];
    float si = 0.0f, st = 0.0f, sd = 0.0f;
#pragma unroll
    for (int t = 0; t < 4; ++t) {
        vi[t]     = *(const float4*)(ri + t * 128 + lane * 4);
        float4 vt = *(const float4*)(rt + t * 128 + lane * 4);
        si += vi[t].x * vi[t].x + vi[t].y * vi[t].y + vi[t].z * vi[t].z + vi[t].w * vi[t].w;
        st += vt.x * vt.x + vt.y * vt.y + vt.z * vt.z + vt.w * vt.w;
        sd += vi[t].x * vt.x + vi[t].y * vt.y + vi[t].z * vt.z + vi[t].w * vt.w;
    }
#pragma unroll
    for (int mask = 16; mask >= 1; mask >>= 1) {
        si += __shfl_xor(si, mask);
        st += __shfl_xor(st, mask);
        sd += __shfl_xor(sd, mask);
    }

    const float inv_i = 1.0f / fmaxf(sqrtf(si), EPS_F);
    if (lane == 0) {
        in_inv[row] = inv_i;
        const float sim = sd * inv_i * (1.0f / fmaxf(sqrtf(st), EPS_F));
        d_pos[row] = sqrtf(fmaxf(2.0f * (1.0f - sim), 1e-12f));
    }

    _Float16* o = inn + (size_t)row * 512;
#pragma unroll
    for (int t = 0; t < 4; ++t) {
        h4 h;
        h.x = (_Float16)(vi[t].x * inv_i);
        h.y = (_Float16)(vi[t].y * inv_i);
        h.z = (_Float16)(vi[t].z * inv_i);
        h.w = (_Float16)(vi[t].w * inv_i);
        *(h4*)(o + t * 128 + lane * 4) = h;
    }
}

// ---------------------------------------------------------------------------
// Kernel 3: WMMA GEMM (inn @ vnn^T) with streaming per-row top-2.
// Workgroup (8 waves) owns a 16-row A tile (staged in LDS; the compiler
// hoists the 16 A fragments into persistent VGPRs). Each wave processes
// groups of 4 column tiles with 4 accumulators. B fragments are double-
// buffered across k-chunks, and sched_barrier(0) fences keep the 8
// global_load_b128 of chunk kc+1 issued BEFORE the 4 WMMAs of chunk kc, so
// the pre-WMMA wait drains only to loadcnt<=8 (one full stage in flight)
// instead of the load->wait(0)->wmma lockstep the scheduler produced before.
// ---------------------------------------------------------------------------
__global__ void __launch_bounds__(256)
mm_gemm_top2_kernel(const _Float16* __restrict__ inn, const _Float16* __restrict__ vnn,
                    int* __restrict__ top2, int B, int V) {
    __shared__ _Float16 As[16 * 512];
    __shared__ float s_v1[8][16], s_v2[8][16];
    __shared__ int   s_c1[8][16], s_c2[8][16];

    const int tid     = threadIdx.x;
    const int wave    = tid >> 5;
    const int lane    = tid & 31;
    const int rowbase = blockIdx.x * 16;

    // Stage the 16x512 f16 A tile into LDS (16 KB), fully coalesced.
    {
        const float4* gsrc = (const float4*)(inn + (size_t)rowbase * 512);
        float4*       ldst = (float4*)As;
#pragma unroll
        for (int i = 0; i < 4; ++i) ldst[tid + i * 256] = gsrc[tid + i * 256];
    }
    __syncthreads();

    const int m  = lane & 15;
    const int hi = lane >> 4;
    const _Float16* pa_base = &As[m * 512 + hi * 8];

    float rv1[8], rv2[8];
    int   rc1[8], rc2[8];
#pragma unroll
    for (int j = 0; j < 8; ++j) { rv1[j] = NEG_INF_F; rv2[j] = NEG_INF_F; rc1[j] = 0; rc2[j] = 0; }

    const int ntiles  = V >> 4;               // 3125
    const int ngroups = (ntiles + 3) >> 2;    // 782

    for (int g = wave; g < ngroups; g += 8) {
        const int t0 = g << 2;

        const _Float16* bp[4];
        int colbase[4];
#pragma unroll
        for (int u = 0; u < 4; ++u) {
            int tt = t0 + u;
            if (tt >= ntiles) tt = ntiles - 1;     // clamp loads; merge skipped below
            colbase[u] = tt << 4;
            bp[u] = vnn + (size_t)(colbase[u] + m) * 512 + hi * 8;
            // Pull the group this wave touches next iteration (g + 8) toward the WGP.
            __builtin_prefetch(bp[u] + (size_t)8 * 4 * 16 * 512, 0, 3);
        }

        v8f acc[4] = {{}, {}, {}, {}};
        AFrag a_buf[2], b_buf[2][4];

        // Prologue: fill stage 0 (k-chunk 0).
        a_buf[0].f4[0] = *(const float4*)(pa_base);
        a_buf[0].f4[1] = *(const float4*)(pa_base + 16);
#pragma unroll
        for (int u = 0; u < 4; ++u) {
            b_buf[0][u].f4[0] = *(const float4*)(bp[u]);
            b_buf[0][u].f4[1] = *(const float4*)(bp[u] + 16);
        }
        __builtin_amdgcn_sched_barrier(0);

#pragma unroll
        for (int kc = 0; kc < 16; ++kc) {
            const int cur = kc & 1;
            const int nxt = cur ^ 1;
            if (kc < 15) {
                const int ko = (kc + 1) * 32;
                a_buf[nxt].f4[0] = *(const float4*)(pa_base + ko);
                a_buf[nxt].f4[1] = *(const float4*)(pa_base + ko + 16);
#pragma unroll
                for (int u = 0; u < 4; ++u) {
                    b_buf[nxt][u].f4[0] = *(const float4*)(bp[u] + ko);
                    b_buf[nxt][u].f4[1] = *(const float4*)(bp[u] + ko + 16);
                }
            }
            // Fence: next-stage loads must all be issued before this stage's
            // WMMAs; WMMAs may not float up into the load block.
            __builtin_amdgcn_sched_barrier(0);
#pragma unroll
            for (int u = 0; u < 4; ++u) {
                acc[u] = __builtin_amdgcn_wmma_f32_16x16x32_f16(
                    false, a_buf[cur].h, false, b_buf[cur][u].h, (short)0, acc[u],
                    false, false);
            }
            __builtin_amdgcn_sched_barrier(0);
        }

        // Per-row top-2 across each tile's 16 columns, fold into running top-2.
#pragma unroll
        for (int u = 0; u < 4; ++u) {
            if (t0 + u < ntiles) {
#pragma unroll
                for (int j = 0; j < 8; ++j) {
                    float v1 = acc[u][j];
                    int   c1 = colbase[u] + m;
                    float v2 = NEG_INF_F;
                    int   c2 = 0;
#pragma unroll
                    for (int mask = 1; mask < 16; mask <<= 1) {
                        const float ov1 = __shfl_xor(v1, mask);
                        const int   oc1 = __shfl_xor(c1, mask);
                        const float ov2 = __shfl_xor(v2, mask);
                        const int   oc2 = __shfl_xor(c2, mask);
                        if (ov1 > v1) {
                            if (ov2 > v1) { v2 = ov2; c2 = oc2; } else { v2 = v1; c2 = c1; }
                            v1 = ov1; c1 = oc1;
                        } else if (ov1 > v2) { v2 = ov1; c2 = oc1; }
                    }
                    if (v1 > rv1[j]) {
                        if (v2 > rv1[j]) { rv2[j] = v2; rc2[j] = c2; }
                        else             { rv2[j] = rv1[j]; rc2[j] = rc1[j]; }
                        rv1[j] = v1; rc1[j] = c1;
                    } else if (v1 > rv2[j]) { rv2[j] = v1; rc2[j] = c1; }
                }
            }
        }
    }

    // Per-wave results to LDS (lanes 0 and 16 hold rows j and j+8).
    if (m == 0) {
#pragma unroll
        for (int j = 0; j < 8; ++j) {
            const int r = j + 8 * hi;
            s_v1[wave][r] = rv1[j]; s_c1[wave][r] = rc1[j];
            s_v2[wave][r] = rv2[j]; s_c2[wave][r] = rc2[j];
        }
    }
    __syncthreads();

    // Final cross-wave merge: 16 threads, one per row.
    if (tid < 16) {
        float v1 = NEG_INF_F, v2 = NEG_INF_F;
        int   c1 = 0, c2 = 0;
        for (int w = 0; w < 8; ++w) {
            const float nv1 = s_v1[w][tid], nv2 = s_v2[w][tid];
            const int   nc1 = s_c1[w][tid], nc2 = s_c2[w][tid];
            if (nv1 > v1) {
                if (nv2 > v1) { v2 = nv2; c2 = nc2; } else { v2 = v1; c2 = c1; }
                v1 = nv1; c1 = nc1;
            } else if (nv1 > v2) { v2 = nv1; c2 = nc1; }
        }
        top2[(rowbase + tid) * 2 + 0] = c1;
        top2[(rowbase + tid) * 2 + 1] = c2;
    }
}

// ---------------------------------------------------------------------------
// Kernel 4: finalize. One wave per row: tie check (vec0 == target), pick the
// negative, recompute its similarity in exact fp32, hinge, atomic accumulate.
// ---------------------------------------------------------------------------
__global__ void __launch_bounds__(256)
mm_finalize_kernel(const float* __restrict__ input, const float* __restrict__ target,
                   const float* __restrict__ veclist, const int* __restrict__ top2,
                   const float* __restrict__ d_pos, const float* __restrict__ in_inv,
                   const float* __restrict__ v_inv, float* __restrict__ out, int B) {
    const int row  = blockIdx.x * 8 + (threadIdx.x >> 5);
    const int lane = threadIdx.x & 31;
    if (row >= B) return;

    const int i0 = top2[row * 2 + 0];
    const int i1 = top2[row * 2 + 1];

    const float* tg = target  + (size_t)row * 512;
    const float* v0 = veclist + (size_t)i0 * 512;
    int mism = 0;
#pragma unroll
    for (int t = 0; t < 4; ++t) {
        const float4 a = *(const float4*)(v0 + t * 128 + lane * 4);
        const float4 b = *(const float4*)(tg + t * 128 + lane * 4);
        mism |= (a.x != b.x) | (a.y != b.y) | (a.z != b.z) | (a.w != b.w);
    }
    const bool eq0 = (__ballot(mism) == 0);
    const int  neg = eq0 ? i1 : i0;

    const float* vn = veclist + (size_t)neg * 512;
    const float* in = input   + (size_t)row * 512;
    float s = 0.0f;
#pragma unroll
    for (int t = 0; t < 4; ++t) {
        const float4 a = *(const float4*)(in + t * 128 + lane * 4);
        const float4 c = *(const float4*)(vn + t * 128 + lane * 4);
        s += a.x * c.x + a.y * c.y + a.z * c.z + a.w * c.w;
    }
#pragma unroll
    for (int mask = 16; mask >= 1; mask >>= 1) s += __shfl_xor(s, mask);

    if (lane == 0) {
        const float sim    = s * in_inv[row] * v_inv[neg];
        const float d_neg  = sqrtf(fmaxf(2.0f * (1.0f - sim), 1e-12f));
        const float margin = 0.5f + d_pos[row] - d_neg;              // GAMMA = 0.5
        const float contrib = 2.0f * fmaxf(margin, 0.0f) / (float)B; // RANK = 2
        atomicAdd(out, contrib);
    }
}

// ---------------------------------------------------------------------------
// Host-side launch
// ---------------------------------------------------------------------------
extern "C" void kernel_launch(void* const* d_in, const int* in_sizes, int n_in,
                              void* d_out, int out_size, void* d_ws, size_t ws_size,
                              hipStream_t stream) {
    const float* input   = (const float*)d_in[0];
    const float* target  = (const float*)d_in[1];
    const float* veclist = (const float*)d_in[2];

    const int D = 512;
    const int B = in_sizes[0] / D;   // 4096
    const int V = in_sizes[2] / D;   // 50000 (multiple of 16)

    // Workspace carve-out (~56 MB total).
    char* p = (char*)d_ws;
    auto carve = [&](size_t bytes) {
        void* q = (void*)p;
        p += (bytes + 255) & ~(size_t)255;
        return q;
    };
    _Float16* vnn   = (_Float16*)carve((size_t)V * D * sizeof(_Float16));
    _Float16* inn   = (_Float16*)carve((size_t)B * D * sizeof(_Float16));
    float*    v_inv = (float*)carve((size_t)V * sizeof(float));
    float*    i_inv = (float*)carve((size_t)B * sizeof(float));
    float*    d_pos = (float*)carve((size_t)B * sizeof(float));
    int*      top2  = (int*)carve((size_t)B * 2 * sizeof(int));
    float*    out   = (float*)d_out;

    mm_zero_kernel<<<1, 1, 0, stream>>>(out);
    mm_normalize_rows_kernel<<<(V + 7) / 8, 256, 0, stream>>>(veclist, vnn, v_inv, V);
    mm_prep_input_kernel<<<(B + 7) / 8, 256, 0, stream>>>(input, target, inn, i_inv, d_pos, B);
    mm_gemm_top2_kernel<<<B / 16, 256, 0, stream>>>(inn, vnn, top2, B, V);
    mm_finalize_kernel<<<(B + 7) / 8, 256, 0, stream>>>(input, target, veclist, top2,
                                                        d_pos, i_inv, v_inv, out, B);
}
